// TemporalEGNN_64020782514256
// MI455X (gfx1250) — matrix operand order repaired and checked by reading, not compiled, not run on hardware
//
#include <hip/hip_runtime.h>
#include <math.h>

// ---------------------------------------------------------------------------
// CDNA5 (gfx1250) wave32 WMMA bf16 implementation of the TemporalEGNN forward.
// Position-update branch of atom_layer is dead code w.r.t. the output -> elided.
// GEMM: 2x2 WMMA tiles per wave, pre-transposed bf16 weights, double-buffered
// LDS pipeline: TDM (tensor_load_to_lds) stages the next weight tile and
// vector loads stage the next A tile while WMMAs consume the current tiles;
// one barrier per K-step; global prefetch two K-tiles ahead.
// ---------------------------------------------------------------------------

typedef __attribute__((ext_vector_type(16))) __bf16 v16bf;
typedef __attribute__((ext_vector_type(8)))  float  v8f;
typedef unsigned uint32x4 __attribute__((ext_vector_type(4)));
typedef int      int32x8  __attribute__((ext_vector_type(8)));
typedef int      int32x4  __attribute__((ext_vector_type(4)));

#ifdef __has_builtin
#  if __has_builtin(__builtin_amdgcn_tensor_load_to_lds)
#    define HAVE_TDM 1
#  endif
#endif
#ifndef HAVE_TDM
#  define HAVE_TDM 0
#endif

constexpr int  kB  = 4, kT = 8, kN = 256, kE = 8192;
constexpr int  kFin = 32, kC = 128, kH = 8, kS = kT + 1;   // seq len 9
constexpr int  kG  = kB * kT;                               // 32 graphs
constexpr long kGN = (long)kG * kN;                         // 8192 node rows
constexpr long kGE = (long)kG * kE;                         // 262144 edge rows

enum { ACT_NONE = 0, ACT_SILU = 1, ACT_SIGMOID = 2, ACT_TANH = 3, ACT_GELU = 4 };

__device__ inline unsigned short f2bf(float f) {
  unsigned u = __float_as_uint(f);
  u += 0x7FFFu + ((u >> 16) & 1u);          // round-to-nearest-even
  return (unsigned short)(u >> 16);
}
__device__ inline float act_apply(float x, int act) {
  switch (act) {
    case ACT_SILU:    return x / (1.f + expf(-x));
    case ACT_SIGMOID: return 1.f / (1.f + expf(-x));
    case ACT_TANH:    return tanhf(x);
    case ACT_GELU:    return 0.5f * x * (1.f + erff(x * 0.70710678118654752f));
    default:          return x;
  }
}

// ---------------------------------------------------------------------------
// GEMM: out[M,N] = act(A[M,K](bf16) @ W[K,N] + bias) + resid
// W supplied TRANSPOSED: Wt[NA, K] (bf16, rows padded to NA = ceil(N/128)*128).
// Block = 256 threads = 8 waves; block tile 64x128; wave tile 32x32 (2x2 WMMA).
// Fragment layouts per CDNA5 ISA 7.12.2.
// ---------------------------------------------------------------------------
constexpr int BM = 64, BN = 128, BK = 32;

union FragAB { v16bf v; unsigned u[8]; uint4 q[2]; };
union FragC  { v8f  v; float    f[8]; };

__global__ __launch_bounds__(256)
void gemm_bf16_kernel(const unsigned short* __restrict__ A,
                      const unsigned short* __restrict__ Wt,
                      const float* __restrict__ bias,
                      const float* __restrict__ resid,
                      float* __restrict__ outF,
                      unsigned short* __restrict__ outB,
                      int M, int K, int N, int NA, int act)
{
  __shared__ unsigned short lA[2][BM * BK];   // A tiles   (2 x 4KB)
  __shared__ unsigned short lB[2][BN * BK];   // Wt tiles  (2 x 8KB)

  const int tid  = threadIdx.x;
  const int lane = tid & 31;
  const int wave = tid >> 5;
  const int wm   = wave >> 2;          // 0..1  (32-row group)
  const int wn   = wave & 3;           // 0..3  (32-col group)
  const int m0   = blockIdx.y * BM;
  const int n0   = blockIdx.x * BN;
  const int half = lane >> 4;          // 0/1
  const int l16  = lane & 15;
  const int nk   = K / BK;

  FragC acc[2][2];
#pragma unroll
  for (int i = 0; i < 2; ++i)
#pragma unroll
    for (int j = 0; j < 2; ++j)
#pragma unroll
      for (int r = 0; r < 8; ++r) acc[i][j].f[r] = 0.f;

  auto stageA = [&](int kt, int b) {
    // A tile: 64x32 bf16, each thread one uint4 (8 bf16)
    int k0 = kt * BK;
    int r  = tid >> 2;
    int cg = (tid & 3) << 3;
    int gm = m0 + r;
    uint4 val = make_uint4(0u, 0u, 0u, 0u);
    if (gm < M) {
      const unsigned short* src = A + (size_t)gm * K + k0 + cg;
      val = *reinterpret_cast<const uint4*>(src);
      __builtin_prefetch(src + BK, 0, 3);   // two K-tiles ahead of compute
    }
    *reinterpret_cast<uint4*>(&lA[b][r * BK + cg]) = val;
  };

  auto stageB = [&](int kt, int b) {
    int k0 = kt * BK;
#if HAVE_TDM
    if (wave == 0) {
      // Tensor DMA: Wt tile [n0..n0+127] x [k0..k0+31] (bf16) -> lB[b].
      // D# per ISA 8.3/8.4: count=1, type=2, data_size=1 (2B),
      // tile 32 x 128, tensor (K x NA), dim0 stride = K.
      unsigned lds_b = (unsigned)(size_t)(void*)&lB[b][0];
      unsigned long long ga =
          (unsigned long long)(size_t)Wt + ((size_t)n0 * K + k0) * 2ull;
      uint32x4 g0;
      g0[0] = 1u;                                   // count=1
      g0[1] = lds_b;                                // lds_addr
      g0[2] = (unsigned)(ga & 0xffffffffu);         // global_addr[31:0]
      g0[3] = (unsigned)((ga >> 32) & 0x1ffffffu) | (2u << 30);  // type=2
      int32x8 g1;
      g1[0] = (int)(1u << 16);                      // data_size=1 -> 2 bytes
      g1[1] = (int)(((unsigned)K & 0xffffu) << 16); // tensor_dim0[15:0]
      g1[2] = (int)((((unsigned)NA & 0xffffu) << 16) |
                    (((unsigned)K >> 16) & 0xffffu));
      g1[3] = (int)((32u << 16) | (((unsigned)NA >> 16) & 0xffffu)); // tile_dim0
      g1[4] = 128;                                  // tile_dim1
      g1[5] = K;                                    // tensor_dim0_stride
      g1[6] = 0;
      g1[7] = 0;
      int32x4 z4 = {0, 0, 0, 0};
#if __clang_major__ >= 23
      int32x8 z8 = {0, 0, 0, 0, 0, 0, 0, 0};
      __builtin_amdgcn_tensor_load_to_lds(g0, g1, z4, z4, z8, 0);
#else
      __builtin_amdgcn_tensor_load_to_lds(g0, g1, z4, z4, 0);
#endif
    }
#else
    { // Wt tile: 128 rows x 32 cols, each thread 2x uint4 (16 bf16)
      int n = tid >> 1;
      int part = (tid & 1) << 4;                    // 0 or 16 elements
      const uint4* src =
          reinterpret_cast<const uint4*>(Wt + (size_t)(n0 + n) * K + k0 + part);
      uint4* dst = reinterpret_cast<uint4*>(&lB[b][n * BK + part]);
      dst[0] = src[0];
      dst[1] = src[1];
    }
#endif
  };

  auto compute = [&](int b) {
    FragAB fa[2], fb[2];
#pragma unroll
    for (int i = 0; i < 2; ++i) {     // A 16x32 fragment (ISA layout)
      const unsigned short* base = &lA[b][(wm * 32 + i * 16 + l16) * BK];
      fa[i].q[0] = *reinterpret_cast<const uint4*>(base + half * 8);
      fa[i].q[1] = *reinterpret_cast<const uint4*>(base + 16 + half * 8);
    }
#pragma unroll
    for (int j = 0; j < 2; ++j) {     // B 32x16 fragment (ISA layout, same k order as A)
      const unsigned short* base =
          &lB[b][(wn * 32 + j * 16 + l16) * BK];
      fb[j].q[0] = *reinterpret_cast<const uint4*>(base + half * 8);
      fb[j].q[1] = *reinterpret_cast<const uint4*>(base + 16 + half * 8);
    }
#pragma unroll
    for (int i = 0; i < 2; ++i)
#pragma unroll
      for (int j = 0; j < 2; ++j)
        acc[i][j].v = __builtin_amdgcn_wmma_f32_16x16x32_bf16(
            false, fa[i].v, false, fb[j].v, (short)0, acc[i][j].v, false, false);
  };

  // ---- software pipeline: stage k+1 while computing k; 1 barrier/K-step ----
  stageA(0, 0);
  stageB(0, 0);
#if HAVE_TDM
  __builtin_amdgcn_s_wait_tensorcnt(0);
#endif
  __syncthreads();
  int buf = 0;
  for (int kt = 0; kt < nk; ++kt) {
    if (kt + 1 < nk) {
      stageA(kt + 1, buf ^ 1);
      stageB(kt + 1, buf ^ 1);
    }
    compute(buf);
#if HAVE_TDM
    __builtin_amdgcn_s_wait_tensorcnt(0);   // next B tile landed (overlapped)
#endif
    __syncthreads();
    buf ^= 1;
  }

#pragma unroll
  for (int i = 0; i < 2; ++i)
#pragma unroll
    for (int j = 0; j < 2; ++j)
#pragma unroll
      for (int r = 0; r < 8; ++r) {   // C/D layout: VGPR r -> row r(+8), N=lane
        int row = m0 + wm * 32 + i * 16 + (half << 3) + r;
        int col = n0 + wn * 32 + j * 16 + l16;
        if (row < M && col < N) {
          float v = acc[i][j].f[r];
          if (bias)  v += bias[col];
          v = act_apply(v, act);
          size_t idx = (size_t)row * N + col;
          if (resid) v += resid[idx];
          if (outF)  outF[idx] = v;
          if (outB)  outB[idx] = f2bf(v);
        }
      }
}

// ---------------------------------------------------------------------------
// Support kernels
// ---------------------------------------------------------------------------
__global__ void fill_f32_kernel(float* p, float v, long n) {
  long i = blockIdx.x * (long)blockDim.x + threadIdx.x;
  if (i < n) p[i] = v;
}
__global__ void cvt_bf16_kernel(const float* __restrict__ s,
                                unsigned short* __restrict__ d, long n) {
  long i = blockIdx.x * (long)blockDim.x + threadIdx.x;
  if (i < n) d[i] = f2bf(s[i]);
}
// weight (Ks,Ns) f32 -> TRANSPOSED (NA,Kp) bf16, zero padded
__global__ void cvt_pad_t_kernel(const float* __restrict__ s,
                                 unsigned short* __restrict__ d,
                                 int Kp, int NA, int Ks, int Ns) {
  long i = blockIdx.x * (long)blockDim.x + threadIdx.x;
  if (i >= (long)NA * Kp) return;
  int n = (int)(i / Kp), k = (int)(i % Kp);
  d[i] = (k < Ks && n < Ns) ? f2bf(s[(size_t)k * Ns + n]) : (unsigned short)0;
}
__global__ void pad_bias_kernel(const float* s, float* d, int Ns, int Np) {
  int i = blockIdx.x * blockDim.x + threadIdx.x;
  if (i < Np) d[i] = (i < Ns) ? s[i] : 0.f;
}
__global__ void count_cols_kernel(const int* __restrict__ ei, float* cnt) {
  long i = blockIdx.x * (long)blockDim.x + threadIdx.x;
  if (i >= kGE) return;
  int g = (int)(i / kE), e = (int)(i % kE);
  int col = ei[(size_t)g * 2 * kE + kE + e];
  atomicAdd(&cnt[(size_t)g * kN + col], 1.f);
}
// sigma input: [e(32) | LnK(h)[col](128)]  -> (GE,160) bf16
__global__ void build_sigma_in_kernel(const unsigned short* __restrict__ ebf,
                                      const unsigned short* __restrict__ knb,
                                      const int* __restrict__ ei,
                                      unsigned short* __restrict__ out) {
  long i = blockIdx.x * (long)blockDim.x + threadIdx.x;
  if (i >= kGE * 160) return;
  int  c  = (int)(i % 160);
  long ge = i / 160;
  int  g  = (int)(ge / kE), e = (int)(ge % kE);
  unsigned short v;
  if (c < 32) v = ebf[ge * 32 + c];
  else {
    int col = ei[(size_t)g * 2 * kE + kE + e];
    v = knb[((size_t)g * kN + col) * kC + (c - 32)];
  }
  out[i] = v;
}
// phi_msg input: [e(32) | LnV(h)[row](128) | LnV(h)[col](128)] -> (GE,288) bf16
__global__ void build_msg_in_kernel(const unsigned short* __restrict__ ebf,
                                    const unsigned short* __restrict__ vnb,
                                    const int* __restrict__ ei,
                                    unsigned short* __restrict__ out) {
  long i = blockIdx.x * (long)blockDim.x + threadIdx.x;
  if (i >= kGE * 288) return;
  int  c  = (int)(i % 288);
  long ge = i / 288;
  int  g  = (int)(ge / kE), e = (int)(ge % kE);
  const int* base = ei + (size_t)g * 2 * kE;
  unsigned short v;
  if (c < 32)       v = ebf[ge * 32 + c];
  else if (c < 160) v = vnb[((size_t)g * kN + base[e]) * kC + (c - 32)];
  else              v = vnb[((size_t)g * kN + base[kE + e]) * kC + (c - 160)];
  out[i] = v;
}
// agg[col] += sigmoid(q[row]*ksig/sqrt(C)) * msg
__global__ void scatter_msg_kernel(const float* __restrict__ qn,
                                   const float* __restrict__ ksig,
                                   const float* __restrict__ msg,
                                   const int* __restrict__ ei,
                                   float* __restrict__ agg) {
  long i = blockIdx.x * (long)blockDim.x + threadIdx.x;
  if (i >= kGE * kC) return;
  int  c  = (int)(i % kC);
  long ge = i / kC;
  int  g  = (int)(ge / kE), e = (int)(ge % kE);
  const int* base = ei + (size_t)g * 2 * kE;
  int row = base[e], col = base[kE + e];
  float q = qn[((size_t)g * kN + row) * kC + c];
  float w = 1.f / (1.f + expf(-q * ksig[i] * 0.08838834764831845f));
  atomicAdd(&agg[((size_t)g * kN + col) * kC + c], w * msg[i]);
}
__global__ void div_cnt_kernel(float* agg, const float* cnt) {
  long i = blockIdx.x * (long)blockDim.x + threadIdx.x;
  if (i >= kGN * kC) return;
  agg[i] /= fmaxf(cnt[i / kC], 1.f);
}
// phi_out input: [agg(128) | h(128)] -> (GN,256) bf16
__global__ void build_phiout_in_kernel(const float* __restrict__ agg,
                                       const float* __restrict__ h,
                                       unsigned short* __restrict__ out) {
  long i = blockIdx.x * (long)blockDim.x + threadIdx.x;
  if (i >= kGN * 256) return;
  int  c  = (int)(i % 256);
  long gn = i / 256;
  float v = (c < kC) ? agg[gn * kC + c] : h[gn * kC + (c - kC)];
  out[i] = f2bf(v);
}
// mean over nodes + sinusoidal time embedding -> seq rows 1..8 per batch
__global__ void pool_embed_kernel(const float* __restrict__ nodeh,
                                  float* __restrict__ seq) {
  int g = blockIdx.x, c = threadIdx.x;
  float s = 0.f;
  for (int n = 0; n < kN; ++n) s += nodeh[((size_t)g * kN + n) * kC + c];
  s *= (1.f / kN);
  int b = g / kT, t = g % kT;
  float f  = expf(-9.210340371976184f * (float)(c & ~1) / (float)kC);
  float pe = (c & 1) ? cosf(t * f) : sinf(t * f);
  seq[((size_t)b * kS + (t + 1)) * kC + c] = s + pe;
}
__global__ void set_cls_kernel(const float* cls, float* seq) {
  int i = blockIdx.x * blockDim.x + threadIdx.x;
  if (i >= kB * kC) return;
  seq[(size_t)(i / kC) * kS * kC + (i % kC)] = cls[i % kC];
}
// tiny MHA: S=9, H=8, dh=16; one block per (b,h)
__global__ void mha9_kernel(const float* __restrict__ qkv, float* __restrict__ o) {
  __shared__ float q[9][16], k[9][16], v[9][16], att[9][9];
  int b = blockIdx.x / kH, h = blockIdx.x % kH, t = threadIdx.x;
  if (t < 144) {
    int i = t / 16, d = t % 16;
    const float* base = qkv + ((size_t)b * kS + i) * (3 * kC);
    q[i][d] = base[h * 16 + d];
    k[i][d] = base[kC + h * 16 + d];
    v[i][d] = base[2 * kC + h * 16 + d];
  }
  __syncthreads();
  if (t < 81) {
    int i = t / 9, j = t % 9;
    float s = 0.f;
    for (int d = 0; d < 16; ++d) s += q[i][d] * k[j][d];
    att[i][j] = s * 0.25f;
  }
  __syncthreads();
  if (t < 9) {
    float mx = -1e30f;
    for (int j = 0; j < 9; ++j) mx = fmaxf(mx, att[t][j]);
    float sm = 0.f, e[9];
    for (int j = 0; j < 9; ++j) { e[j] = expf(att[t][j] - mx); sm += e[j]; }
    for (int j = 0; j < 9; ++j) att[t][j] = e[j] / sm;
  }
  __syncthreads();
  if (t < 144) {
    int i = t / 16, d = t % 16;
    float s = 0.f;
    for (int j = 0; j < 9; ++j) s += att[i][j] * v[j][d];
    o[((size_t)b * kS + i) * kC + h * 16 + d] = s;
  }
}
// out = LayerNorm(x + r) * g + b  (width 128, one block per row)
__global__ void add_ln_kernel(const float* __restrict__ x,
                              const float* __restrict__ r,
                              const float* __restrict__ g,
                              const float* __restrict__ bb,
                              float* __restrict__ out) {
  __shared__ float red[kC];
  int row = blockIdx.x, c = threadIdx.x;
  float v = x[(size_t)row * kC + c] + (r ? r[(size_t)row * kC + c] : 0.f);
  red[c] = v; __syncthreads();
  for (int s = 64; s > 0; s >>= 1) { if (c < s) red[c] += red[c + s]; __syncthreads(); }
  float mean = red[0] / kC; __syncthreads();
  float d = v - mean;
  red[c] = d * d; __syncthreads();
  for (int s = 64; s > 0; s >>= 1) { if (c < s) red[c] += red[c + s]; __syncthreads(); }
  float var = red[0] / kC;
  out[(size_t)row * kC + c] = d * rsqrtf(var + 1e-5f) * g[c] + bb[c];
}
__global__ void gather_h0_kernel(const float* __restrict__ nodeh,
                                 float* __restrict__ h0f,
                                 unsigned short* __restrict__ h0b) {
  long i = blockIdx.x * (long)blockDim.x + threadIdx.x;
  if (i >= (long)kB * kN * kC) return;
  int b = (int)(i / ((long)kN * kC));
  long rem = i % ((long)kN * kC);
  float v = nodeh[((size_t)(b * kT) * kN) * kC + rem];  // t = 0 graph
  h0f[i] = v; h0b[i] = f2bf(v);
}
__global__ void zcls_kernel(const float* seq, unsigned short* zb) {
  int i = blockIdx.x * blockDim.x + threadIdx.x;
  if (i >= kB * kC) return;
  zb[i] = f2bf(seq[(size_t)(i / kC) * kS * kC + (i % kC)]);
}
// decoder softmax weights over N; emits vw = bf16(V_z[b] * wts[b,n])
__global__ void dec_wts_kernel(const float* __restrict__ Q,
                               const float* __restrict__ kz,
                               const float* __restrict__ vz,
                               unsigned short* __restrict__ vw) {
  __shared__ float red[kN];
  int b = blockIdx.x, n = threadIdx.x;
  float s = 0.f;
  for (int c = 0; c < kC; ++c) s += Q[((size_t)b * kN + n) * kC + c] * kz[b * kC + c];
  s *= 0.08838834764831845f;
  red[n] = s; __syncthreads();
  for (int st = 128; st > 0; st >>= 1) { if (n < st) red[n] = fmaxf(red[n], red[n + st]); __syncthreads(); }
  float mx = red[0]; __syncthreads();
  float e = expf(s - mx);
  red[n] = e; __syncthreads();
  for (int st = 128; st > 0; st >>= 1) { if (n < st) red[n] += red[n + st]; __syncthreads(); }
  float w = e / red[0];
  for (int c = 0; c < kC; ++c)
    vw[((size_t)b * kN + n) * kC + c] = f2bf(vz[b * kC + c] * w);
}
__global__ void final_pred_kernel(const float* __restrict__ pos,
                                  const float* __restrict__ posd,
                                  float* __restrict__ out) {
  long i = blockIdx.x * (long)blockDim.x + threadIdx.x;
  if (i >= (long)kB * kN * 3) return;
  int b = (int)(i / (kN * 3));
  int r = (int)(i % (kN * 3));
  int n = r / 3, k = r % 3;
  out[i] = pos[((size_t)b * kT * kN + n) * 3 + k] + posd[((size_t)b * kN + n) * 16 + k];
}

// ---------------------------------------------------------------------------
// Host orchestration
// ---------------------------------------------------------------------------
struct Ws {
  char* base; size_t off;
  void* take(size_t bytes) {
    off = (off + 255) & ~(size_t)255;
    void* p = base + off; off += bytes; return p;
  }
};
static inline unsigned gcd1(long n) { return (unsigned)((n + 255) / 256); }
static inline int nalloc(int N) { return ((N + 127) / 128) * 128; }

static void gemm(hipStream_t s, const unsigned short* A, const unsigned short* Wt,
                 const float* bias, const float* resid,
                 float* outF, unsigned short* outB, long M, int K, int N, int act) {
  dim3 grid((N + BN - 1) / BN, (unsigned)((M + BM - 1) / BM));
  gemm_bf16_kernel<<<grid, 256, 0, s>>>(A, Wt, bias, resid, outF, outB,
                                        (int)M, K, N, nalloc(N), act);
}
// convert weight (Ks,Ns) f32 -> transposed padded (NA,Kp) bf16
static unsigned short* cvt_w(hipStream_t s, Ws& ws, const void* w,
                             int Ks, int Ns, int Kp, int Np) {
  int NA = nalloc(Np);
  unsigned short* d = (unsigned short*)ws.take((size_t)NA * Kp * 2);
  long t = (long)NA * Kp;
  cvt_pad_t_kernel<<<gcd1(t), 256, 0, s>>>((const float*)w, d, Kp, NA, Ks, Ns);
  return d;
}

extern "C" void kernel_launch(void* const* d_in, const int* in_sizes, int n_in,
                              void* d_out, int out_size, void* d_ws, size_t ws_size,
                              hipStream_t stream) {
  (void)in_sizes; (void)n_in; (void)out_size; (void)ws_size;
  const float* x   = (const float*)d_in[0];
  const float* pos = (const float*)d_in[1];
  const int*   ei  = (const int*)  d_in[2];
  const float* ea  = (const float*)d_in[3];
  // d_in[4] edge_vec, d_in[5] pos_frac: dead code w.r.t. output

  // ---- param pytree leaf indices (jax sorted-key flatten order) ----
  auto P = [&](int i) { return (const float*)d_in[i]; };
  const int CLS = 6;
  const int D0 = 7;      // dec: ff1.b ff1.w ff2.b ff2.w k.b k.w ln1.b ln1.g
                         //      ln2.b ln2.g o.b o.w pos.b pos.w q.b q.w v.b v.w
  const int LYR0 = 25;   // 5 layers x 32 leaves
  const int LIN_IN_B = 185, LIN_IN_W = 186, LIN_OUT_B = 187, LIN_OUT_W = 188;
  const int TL0 = 189;   // 4 tlayers x 12 leaves

  Ws ws{(char*)d_ws, 0};
  hipStream_t s = stream;

  // ---- weight conversion (f32 -> transposed padded bf16) ----
  unsigned short *wLnE[5], *wLnK[5], *wLnQ[5], *wLnV[5];
  unsigned short *wSig0[5], *wSig1[5], *wMsg0[5], *wMsg1[5], *wOut0[5], *wOut1[5];
  for (int l = 0; l < 5; ++l) {
    int L = LYR0 + l * 32;
    wLnE[l]  = cvt_w(s, ws, P(L + 1),  32, 32, 32, 32);
    wLnK[l]  = cvt_w(s, ws, P(L + 3), 128, 128, 128, 128);
    wLnQ[l]  = cvt_w(s, ws, P(L + 5), 128, 128, 128, 128);
    wLnV[l]  = cvt_w(s, ws, P(L + 7), 128, 128, 128, 128);
    wMsg0[l] = cvt_w(s, ws, P(L + 21), 288, 128, 288, 128);
    wMsg1[l] = cvt_w(s, ws, P(L + 23), 128, 128, 128, 128);
    wOut0[l] = cvt_w(s, ws, P(L + 25), 256, 128, 256, 128);
    wOut1[l] = cvt_w(s, ws, P(L + 27), 128, 128, 128, 128);
    wSig0[l] = cvt_w(s, ws, P(L + 29), 160, 128, 160, 128);
    wSig1[l] = cvt_w(s, ws, P(L + 31), 128, 128, 128, 128);
  }
  unsigned short* wLinIn  = cvt_w(s, ws, P(LIN_IN_W),  32, 128, 32, 128);
  unsigned short* wLinOut = cvt_w(s, ws, P(LIN_OUT_W), 128, 128, 128, 128);
  unsigned short *wInW[4], *wTOut[4], *wTF1[4], *wTF2[4];
  for (int t = 0; t < 4; ++t) {
    int Tb = TL0 + t * 12;
    wTF1[t]  = cvt_w(s, ws, P(Tb + 1), 128, 512, 128, 512);
    wTF2[t]  = cvt_w(s, ws, P(Tb + 3), 512, 128, 512, 128);
    wInW[t]  = cvt_w(s, ws, P(Tb + 5), 128, 384, 128, 384);
    wTOut[t] = cvt_w(s, ws, P(Tb + 11), 128, 128, 128, 128);
  }
  unsigned short* wDf1 = cvt_w(s, ws, P(D0 + 1), 128, 128, 128, 128);
  unsigned short* wDf2 = cvt_w(s, ws, P(D0 + 3), 128, 128, 128, 128);
  unsigned short* wDk  = cvt_w(s, ws, P(D0 + 5), 128, 128, 128, 128);
  unsigned short* wDo  = cvt_w(s, ws, P(D0 + 11), 128, 128, 128, 128);
  unsigned short* wDp  = cvt_w(s, ws, P(D0 + 13), 128, 3, 128, 16);
  unsigned short* wDq  = cvt_w(s, ws, P(D0 + 15), 128, 128, 128, 128);
  unsigned short* wDv  = cvt_w(s, ws, P(D0 + 17), 128, 128, 128, 128);
  float* posBias = (float*)ws.take(16 * 4);
  pad_bias_kernel<<<1, 16, 0, s>>>(P(D0 + 12), posBias, 3, 16);

  // ---- activation buffers ----
  float* cnt   = (float*)ws.take(kGN * 4);
  float* h     = (float*)ws.take(kGN * kC * 4);
  float* h2    = (float*)ws.take(kGN * kC * 4);
  float* nodeh = (float*)ws.take(kGN * kC * 4);
  unsigned short* hbf  = (unsigned short*)ws.take(kGN * kC * 2);
  unsigned short* xbf  = (unsigned short*)ws.take(kGN * kFin * 2);
  unsigned short* eabf = (unsigned short*)ws.take(kGE * 32 * 2);
  unsigned short* ebf  = (unsigned short*)ws.take(kGE * 32 * 2);
  float* qn = (float*)ws.take(kGN * kC * 4);
  unsigned short* knb = (unsigned short*)ws.take(kGN * kC * 2);
  unsigned short* vnb = (unsigned short*)ws.take(kGN * kC * 2);
  unsigned short* edgeIn  = (unsigned short*)ws.take(kGE * 288 * 2);
  unsigned short* edgeMid = (unsigned short*)ws.take(kGE * kC * 2);
  float* ksig = (float*)ws.take(kGE * kC * 4);
  float* msg  = (float*)ws.take(kGE * kC * 4);
  float* agg  = (float*)ws.take(kGN * kC * 4);
  unsigned short* phiIn  = (unsigned short*)ws.take(kGN * 256 * 2);
  unsigned short* phiMid = (unsigned short*)ws.take(kGN * kC * 2);
  const long SRows = kB * kS;  // 36
  float* seq  = (float*)ws.take(SRows * kC * 4);
  float* seq2 = (float*)ws.take(SRows * kC * 4);
  float* qkv  = (float*)ws.take(SRows * 3 * kC * 4);
  float* attn = (float*)ws.take(SRows * kC * 4);
  float* attnO = (float*)ws.take(SRows * kC * 4);
  float* ffo  = (float*)ws.take(SRows * kC * 4);
  unsigned short* seqb  = (unsigned short*)ws.take(SRows * kC * 2);
  unsigned short* attnb = (unsigned short*)ws.take(SRows * kC * 2);
  unsigned short* ffmb  = (unsigned short*)ws.take(SRows * 512 * 2);
  const long DR = (long)kB * kN;  // 1024
  float* h0f = (float*)ws.take(DR * kC * 4);
  unsigned short* h0b = (unsigned short*)ws.take(DR * kC * 2);
  unsigned short* zb  = (unsigned short*)ws.take(kB * kC * 2);
  float* kz = (float*)ws.take(kB * kC * 4);
  float* vz = (float*)ws.take(kB * kC * 4);
  float* Qd = (float*)ws.take(DR * kC * 4);
  unsigned short* vwb = (unsigned short*)ws.take(DR * kC * 2);
  float* attnD = (float*)ws.take(DR * kC * 4);
  float* x1 = (float*)ws.take(DR * kC * 4);
  float* x2 = (float*)ws.take(DR * kC * 4);
  unsigned short* x1b = (unsigned short*)ws.take(DR * kC * 2);
  unsigned short* x2b = (unsigned short*)ws.take(DR * kC * 2);
  unsigned short* dffb = (unsigned short*)ws.take(DR * kC * 2);
  float* dffo = (float*)ws.take(DR * kC * 4);
  float* posd = (float*)ws.take(DR * 16 * 4);

  // ---- EGNN ----
  fill_f32_kernel<<<gcd1(kGN), 256, 0, s>>>(cnt, 0.f, kGN);
  count_cols_kernel<<<gcd1(kGE), 256, 0, s>>>(ei, cnt);
  cvt_bf16_kernel<<<gcd1(kGN * kFin), 256, 0, s>>>(x, xbf, kGN * kFin);
  gemm(s, xbf, wLinIn, P(LIN_IN_B), nullptr, h, hbf, kGN, 32, 128, ACT_NONE);
  cvt_bf16_kernel<<<gcd1(kGE * 32), 256, 0, s>>>(ea, eabf, kGE * 32);

  for (int l = 0; l < 5; ++l) {
    int L = LYR0 + l * 32;
    gemm(s, eabf, wLnE[l], P(L + 0), nullptr, nullptr, ebf, kGE, 32, 32, ACT_NONE);
    gemm(s, hbf, wLnQ[l], P(L + 4), nullptr, qn, nullptr, kGN, 128, 128, ACT_NONE);
    gemm(s, hbf, wLnK[l], P(L + 2), nullptr, nullptr, knb, kGN, 128, 128, ACT_NONE);
    gemm(s, hbf, wLnV[l], P(L + 6), nullptr, nullptr, vnb, kGN, 128, 128, ACT_NONE);
    build_sigma_in_kernel<<<gcd1(kGE * 160), 256, 0, s>>>(ebf, knb, ei, edgeIn);
    gemm(s, edgeIn, wSig0[l], P(L + 28), nullptr, nullptr, edgeMid, kGE, 160, 128, ACT_SILU);
    gemm(s, edgeMid, wSig1[l], P(L + 30), nullptr, ksig, nullptr, kGE, 128, 128, ACT_SILU);
    build_msg_in_kernel<<<gcd1(kGE * 288), 256, 0, s>>>(ebf, vnb, ei, edgeIn);
    gemm(s, edgeIn, wMsg0[l], P(L + 20), nullptr, nullptr, edgeMid, kGE, 288, 128, ACT_SILU);
    gemm(s, edgeMid, wMsg1[l], P(L + 22), nullptr, msg, nullptr, kGE, 128, 128, ACT_SILU);
    fill_f32_kernel<<<gcd1(kGN * kC), 256, 0, s>>>(agg, 0.f, kGN * kC);
    scatter_msg_kernel<<<gcd1(kGE * kC), 256, 0, s>>>(qn, ksig, msg, ei, agg);
    div_cnt_kernel<<<gcd1(kGN * kC), 256, 0, s>>>(agg, cnt);
    build_phiout_in_kernel<<<gcd1(kGN * 256), 256, 0, s>>>(agg, h, phiIn);
    gemm(s, phiIn, wOut0[l], P(L + 24), nullptr, nullptr, phiMid, kGN, 256, 128, ACT_SILU);
    gemm(s, phiMid, wOut1[l], P(L + 26), h /*resid*/, h2, hbf, kGN, 128, 128, ACT_NONE);
    float* tmp = h; h = h2; h2 = tmp;
  }
  gemm(s, hbf, wLinOut, P(LIN_OUT_B), nullptr, nodeh, nullptr, kGN, 128, 128, ACT_NONE);

  // ---- transformer over (B, 9, C) ----
  pool_embed_kernel<<<kG, kC, 0, s>>>(nodeh, seq);
  set_cls_kernel<<<gcd1(kB * kC), 256, 0, s>>>(P(CLS), seq);
  for (int t = 0; t < 4; ++t) {
    int Tb = TL0 + t * 12;
    cvt_bf16_kernel<<<gcd1(SRows * kC), 256, 0, s>>>(seq, seqb, SRows * kC);
    gemm(s, seqb, wInW[t], P(Tb + 4), nullptr, qkv, nullptr, SRows, 128, 384, ACT_NONE);
    mha9_kernel<<<kB * kH, 160, 0, s>>>(qkv, attn);
    cvt_bf16_kernel<<<gcd1(SRows * kC), 256, 0, s>>>(attn, attnb, SRows * kC);
    gemm(s, attnb, wTOut[t], P(Tb + 10), nullptr, attnO, nullptr, SRows, 128, 128, ACT_NONE);
    add_ln_kernel<<<SRows, kC, 0, s>>>(seq, attnO, P(Tb + 7), P(Tb + 6), seq2);
    cvt_bf16_kernel<<<gcd1(SRows * kC), 256, 0, s>>>(seq2, seqb, SRows * kC);
    gemm(s, seqb, wTF1[t], P(Tb + 0), nullptr, nullptr, ffmb, SRows, 128, 512, ACT_GELU);
    gemm(s, ffmb, wTF2[t], P(Tb + 2), nullptr, ffo, nullptr, SRows, 512, 128, ACT_NONE);
    add_ln_kernel<<<SRows, kC, 0, s>>>(seq2, ffo, P(Tb + 9), P(Tb + 8), seq);
  }

  // ---- decoder ----
  gather_h0_kernel<<<gcd1(DR * kC), 256, 0, s>>>(nodeh, h0f, h0b);
  zcls_kernel<<<gcd1(kB * kC), 256, 0, s>>>(seq, zb);
  gemm(s, zb, wDk, P(D0 + 4), nullptr, kz, nullptr, kB, 128, 128, ACT_NONE);
  gemm(s, zb, wDv, P(D0 + 16), nullptr, vz, nullptr, kB, 128, 128, ACT_NONE);
  gemm(s, h0b, wDq, P(D0 + 14), nullptr, Qd, nullptr, DR, 128, 128, ACT_NONE);
  dec_wts_kernel<<<kB, kN, 0, s>>>(Qd, kz, vz, vwb);
  gemm(s, vwb, wDo, P(D0 + 10), nullptr, attnD, nullptr, DR, 128, 128, ACT_NONE);
  add_ln_kernel<<<(unsigned)DR, kC, 0, s>>>(h0f, attnD, P(D0 + 7), P(D0 + 6), x1);
  cvt_bf16_kernel<<<gcd1(DR * kC), 256, 0, s>>>(x1, x1b, DR * kC);
  gemm(s, x1b, wDf1, P(D0 + 0), nullptr, nullptr, dffb, DR, 128, 128, ACT_SILU);
  gemm(s, dffb, wDf2, P(D0 + 2), nullptr, dffo, nullptr, DR, 128, 128, ACT_NONE);
  add_ln_kernel<<<(unsigned)DR, kC, 0, s>>>(x1, dffo, P(D0 + 9), P(D0 + 8), x2);
  cvt_bf16_kernel<<<gcd1(DR * kC), 256, 0, s>>>(x2, x2b, DR * kC);
  gemm(s, x2b, wDp, posBias, nullptr, posd, nullptr, DR, 128, 16, ACT_NONE);
  final_pred_kernel<<<gcd1(DR * 3), 256, 0, s>>>(pos, posd, (float*)d_out);
}
